// SpikingConv3DLayer_8246337208357
// MI455X (gfx1250) — matrix operand order repaired and checked by reading, not compiled, not run on hardware
//
#include <hip/hip_runtime.h>
#include <hip/hip_bf16.h>

// ---------------------------------------------------------------------------
// SpikingConv3DLayer fused kernel for gfx1250 (MI455X).
//  grid = (H, B); block = 256 threads = 8 waves (wave32).
//  Each workgroup owns one (b, h) row: 32 pixels x 64 output channels.
//  Wave (mt,nt) owns a 16x16 WMMA C-tile. Membrane state lives in 8 VGPRs
//  per lane for all 50 timesteps; conv_x never touches HBM. Spike output is
//  streamed to HBM with the Tensor Data Mover (tensor_store_from_lds),
//  double-buffered in LDS and tracked with TENSORcnt.
// ---------------------------------------------------------------------------

static constexpr int NB   = 8;
static constexpr int CI   = 32;
static constexpr int CO   = 64;
static constexpr int NT   = 50;
static constexpr int NH   = 32;
static constexpr int NW   = 32;

typedef __attribute__((ext_vector_type(16))) __bf16 v16bf;
typedef __attribute__((ext_vector_type(8)))  __bf16 v8bf;
typedef __attribute__((ext_vector_type(8)))  float  v8f;
typedef __attribute__((ext_vector_type(4)))  unsigned int v4u;
typedef __attribute__((ext_vector_type(8)))  unsigned int v8u;

union Frag { v16bf v; v8bf h[2]; unsigned short u[16]; };

__device__ __forceinline__ unsigned short f2bf(float f) {
  unsigned u = __builtin_bit_cast(unsigned, f);
  u += 0x7fffu + ((u >> 16) & 1u);            // round-to-nearest-even
  return (unsigned short)(u >> 16);
}

__device__ __forceinline__ v8bf ld8(const unsigned short* p) {
  return *(const v8bf*)p;                     // 16B LDS load (index kept 8-aligned)
}

__device__ __forceinline__ v8f wmma_bf16(v16bf a, v16bf b, v8f c) {
  // D = A(16x32 bf16) x B(32x16 bf16) + C(16x16 f32)
  return __builtin_amdgcn_wmma_f32_16x16x32_bf16(
      /*neg_a=*/false, a, /*neg_b=*/false, b,
      /*c_mod=*/(short)0, c, /*reuse_a=*/false, /*reuse_b=*/false);
}

// TDM: async store of one (64 ch x 32 w) f32 tile, LDS -> out[b,:,t,h,:].
// 2D descriptor => groups 0/1 only (VADDR2/VADDR3 NULL).
__device__ __forceinline__ void tdm_store_row(const float* out, int b, int h,
                                              int t, const float* ldsbuf) {
  const unsigned ldsOff = (unsigned)(size_t)(const void*)ldsbuf; // LDS byte offset
  const unsigned long long ga =
      (unsigned long long)(size_t)out +
      (((size_t)b * CO * NT + t) * (NH * NW) + (size_t)h * NW) * sizeof(float);
  v4u g0;
  g0[0] = 1u;                                  // count=1, user descriptor
  g0[1] = ldsOff;                              // lds_addr
  g0[2] = (unsigned)ga;                        // global_addr[31:0]
  g0[3] = ((unsigned)(ga >> 32) & 0x01FFFFFFu) // global_addr[56:32]
          | 0x80000000u;                       // type=2 ("image")
  v8u g1;
  g1[0] = 0x00020000u;                         // wg_mask=0, data_size=2 (4B)
  g1[1] = (unsigned)NW << 16;                  // tensor_dim0 = 32
  g1[2] = (unsigned)CO << 16;                  // tensor_dim1 = 64
  g1[3] = (unsigned)NW << 16;                  // tile_dim0 = 32
  g1[4] = (unsigned)CO;                        // tile_dim1 = 64, tile_dim2 = 0
  g1[5] = (unsigned)(NT * NH * NW);            // tensor_dim0_stride = 51200
  g1[6] = 0u;                                  // stride hi / dim1_stride lo
  g1[7] = 0u;
  asm volatile("tensor_store_from_lds %0, %1" :: "s"(g0), "s"(g1) : "memory");
}

// ---------------------------------------------------------------------------
// Prep: gram matrix d[a][f] = <w_a, w_f> over 864 taps (bf16), inv_norm[a].
// ---------------------------------------------------------------------------
__global__ void prep_kernel(const float* __restrict__ w,
                            unsigned short* __restrict__ d_bf,
                            float* __restrict__ inv_norm) {
  const int tid = threadIdx.x;
  for (int e = tid; e < CO * CO; e += 256) {
    const int a = e >> 6, f = e & 63;
    const float* wa = w + (size_t)a * 864;
    const float* wf = w + (size_t)f * 864;
    float acc = 0.0f;
    for (int k = 0; k < 864; ++k) acc = fmaf(wa[k], wf[k], acc);
    d_bf[e] = f2bf(acc);
    if (a == f) inv_norm[a] = 1.0f / (acc + 1e-8f);
  }
}

// ---------------------------------------------------------------------------
// Fused conv3d + spiking recurrence.
// ---------------------------------------------------------------------------
__global__ void __launch_bounds__(256)
spiking_kernel(const float* __restrict__ x, const float* __restrict__ w,
               const float* __restrict__ beta_p, const float* __restrict__ bias_p,
               const unsigned short* __restrict__ d_bf,
               const float* __restrict__ inv_norm,
               float* __restrict__ out) {
  // LDS: 18KB + 108KB + 8KB + 4KB + 16KB = 154KB
  __shared__ alignas(16) unsigned short xs[9 * NW * CI];    // [dt*3+dh][w][ci] bf16
  __shared__ alignas(16) unsigned short wB[27 * CO * CI];   // [tap][co][ci]   bf16
  __shared__ alignas(16) unsigned short dTl[CO * CO];       // [e][bch] (symmetric)
  __shared__ alignas(16) unsigned short spkl[NW * CO];      // [pixel][ch]     bf16
  __shared__ alignas(16) float spkf[2][CO * NW];            // [ch][w] f32, TDM src

  const int tid  = threadIdx.x;
  const int wave = tid >> 5;
  const int lane = tid & 31;
  const int half = lane >> 4;     // which 16-lane half
  const int l16  = lane & 15;
  const int mt   = wave & 1;      // pixel tile (2 x 16 pixels)
  const int nt   = wave >> 1;     // channel tile (4 x 16 channels)
  const int h    = blockIdx.x;
  const int b    = blockIdx.y;

  const int pixRow = mt * 16 + l16;   // A-matrix row = pixel
  const int ch     = nt * 16 + l16;   // C/D column   = output channel
  const float beta = beta_p[0];
  const float ivn  = inv_norm[ch];
  const float bia  = bias_p[ch];

  // --- one-time staging: weights in B layout, gram matrix, spike init ---
  for (int p = tid; p < CO * CI; p += 256) {
    const int co = p >> 5, ci = p & 31;
    const float* wp = w + ((size_t)co * CI + ci) * 27;
    for (int tap = 0; tap < 27; ++tap)
      wB[(tap * CO + co) * CI + ci] = f2bf(wp[tap]);
  }
  for (int p = tid; p < CO * CO; p += 256) dTl[p] = d_bf[p];
  for (int p = tid; p < NW * CO; p += 256) spkl[p] = 0;
  __syncthreads();

  v8f memv = {};   // membrane potential, persistent across t

  for (int t = 0; t < NT; ++t) {
    // ---- stage x halo slices (f32 -> bf16, ci-contiguous for A frags) ----
    {
      const int wq  = tid & 31;
      const int cib = tid >> 5;          // 0..7
      for (int dt = 0; dt < 3; ++dt) {
        const int tt = t + dt - 1;
        if (tt < 0 || tt >= NT) continue;
        for (int dh = 0; dh < 3; ++dh) {
          const int hh = h + dh - 1;
          if (hh < 0 || hh >= NH) continue;
          for (int j = 0; j < 4; ++j) {
            const int ci = j * 8 + cib;
            const float* xp =
                x + (((size_t)(b * CI + ci) * NT + tt) * NH + hh) * NW;
            xs[((dt * 3 + dh) * NW + wq) * CI + ci] = f2bf(xp[wq]);
            if (j == 0 && tt + 1 < NT)
              __builtin_prefetch(xp + NH * NW, 0, 0);   // next t-slice (L2)
          }
        }
      }
    }
    __syncthreads();   // (A) xs ready; prev-step spkl writes ordered

    // ---- conv: inp[pixel,co] = sum_taps X_tap[pixel,ci] * W_tap[ci,co] ----
    v8f acc = {};
    for (int dt = 0; dt < 3; ++dt) {
      const int tt = t + dt - 1;
      if (tt < 0 || tt >= NT) continue;
      for (int dh = 0; dh < 3; ++dh) {
        const int hh = h + dh - 1;
        if (hh < 0 || hh >= NH) continue;
#pragma unroll
        for (int dw = 0; dw < 3; ++dw) {
          const int tap = (dt * 3 + dh) * 3 + dw;
          Frag a;
          const int wc = pixRow + dw - 1;        // shifted pixel column
          if (wc >= 0 && wc < NW) {
            const unsigned short* xr = &xs[((dt * 3 + dh) * NW + wc) * CI];
            a.h[0] = ld8(xr + 16 * half);        // K = 16*half + 0..7
            a.h[1] = ld8(xr + 16 * half + 8);    // K = 16*half + 8..15
          } else {
#pragma unroll
            for (int i = 0; i < 16; ++i) a.u[i] = 0;   // zero-pad row
          }
          Frag bb;
          const unsigned short* wr = &wB[(tap * CO + ch) * CI + 16 * half];
          bb.h[0] = ld8(wr);                     // K = 16*half + 0..7
          bb.h[1] = ld8(wr + 8);                 // K = 16*half + 8..15
          acc = wmma_bf16(a.v, bb.v, acc);
        }
      }
    }

    // ---- rst[pixel,e] = sum_b spk_prev[pixel,b] * d[b,e]  (K = 64) ----
    v8f rst = {};
#pragma unroll
    for (int kc = 0; kc < 2; ++kc) {
      Frag a;
      const unsigned short* sr = &spkl[pixRow * CO + kc * 32];
      a.h[0] = ld8(sr + 16 * half);
      a.h[1] = ld8(sr + 16 * half + 8);
      Frag bb;
      const unsigned short* dr = &dTl[ch * CO + kc * 32 + 16 * half];
      bb.h[0] = ld8(dr);
      bb.h[1] = ld8(dr + 8);
      rst = wmma_bf16(a.v, bb.v, rst);
    }

    // ---- membrane update + threshold ----
    memv = (memv - rst) * beta + acc * (1.0f - beta);
    float sp[8];
#pragma unroll
    for (int r = 0; r < 8; ++r)
      sp[r] = (memv[r] * ivn - bia) > 0.0f ? 1.0f : 0.0f;

    // Gate reuse of spkf[t&1]: the TDM store issued at step t-2 must be done.
    if (wave == 0) __builtin_amdgcn_s_wait_tensorcnt(1);
    __syncthreads();   // (B) spkl reads done + tensorcnt gate for everyone

    // spikes -> recurrence buffer (bf16) and TDM staging tile (f32, [ch][w])
#pragma unroll
    for (int r = 0; r < 8; ++r)
      spkl[(mt * 16 + 8 * half + r) * CO + ch] =
          sp[r] > 0.0f ? (unsigned short)0x3F80 : (unsigned short)0;
    {
      float* dst = &spkf[t & 1][ch * NW + mt * 16 + 8 * half];
      *(float4*)dst       = make_float4(sp[0], sp[1], sp[2], sp[3]);
      *(float4*)(dst + 4) = make_float4(sp[4], sp[5], sp[6], sp[7]);
    }
    __syncthreads();   // (C) spike buffers fully written

    // One async DMA per (WG, t): 64x32 f32 tile -> out[b, :, t, h, :]
    if (wave == 0) tdm_store_row(out, b, h, t, &spkf[t & 1][0]);
  }

  // Drain outstanding tensor stores before the workgroup's LDS can go away
  // (S_ENDPGM also does an implicit wait-idle).
  if (wave == 0) __builtin_amdgcn_s_wait_tensorcnt(0);
}

extern "C" void kernel_launch(void* const* d_in, const int* in_sizes, int n_in,
                              void* d_out, int out_size, void* d_ws, size_t ws_size,
                              hipStream_t stream) {
  (void)in_sizes; (void)n_in; (void)out_size; (void)ws_size;
  const float* x    = (const float*)d_in[0];
  const float* w    = (const float*)d_in[1];
  const float* beta = (const float*)d_in[2];
  const float* bias = (const float*)d_in[3];

  unsigned short* d_bf = (unsigned short*)d_ws;                       // 8KB
  float* inv_norm = (float*)((char*)d_ws + CO * CO * sizeof(unsigned short));

  prep_kernel<<<1, 256, 0, stream>>>(w, d_bf, inv_norm);
  spiking_kernel<<<dim3(NH, NB), 256, 0, stream>>>(
      x, w, beta, bias, d_bf, inv_norm, (float*)d_out);
}
